// RNN_6837587936220
// MI455X (gfx1250) — compile-verified, matrix operand order checked by reading, not run on hardware
//
#include <hip/hip_runtime.h>

#define T_LEN 2048
#define Q_LEN 64
#define A_LEN 32
#define E_DIM 300
#define E_PAD 320
#define HID   512
#define G4    2048   /* 4*H */
#define H2    1024   /* 2*H */
#define NWG   16     /* workgroups per direction for the text scan */

typedef __attribute__((ext_vector_type(16))) __bf16 v16bf;
typedef __attribute__((ext_vector_type(8)))  __bf16 v8bf;
typedef __attribute__((ext_vector_type(8)))  float  v8f;

__device__ __forceinline__ unsigned short f2bf(float f) {
    unsigned int u = __float_as_uint(f);
    u += 0x7FFFu + ((u >> 16) & 1u);          // round-to-nearest-even
    return (unsigned short)(u >> 16);
}

__device__ __forceinline__ float sigf(float x) { return 1.f / (1.f + __expf(-x)); }

// CDNA5 async global->LDS copy (ASYNCcnt path). lds_off = byte offset of the
// LDS destination (low 32 bits of a generic shared pointer), gaddr = global VA.
__device__ __forceinline__ void async_load_b128(unsigned lds_off, const void* gptr) {
    asm volatile("global_load_async_to_lds_b128 %0, %1, off"
                 :: "v"(lds_off), "v"((unsigned long long)(size_t)gptr)
                 : "memory");
}
__device__ __forceinline__ void wait_async0() {
    asm volatile("s_wait_asynccnt 0x0" ::: "memory");
}

// ---------------------------------------------------------------------------
// Embedding gather + f32->bf16, K padded 300->320 with zeros; rows>=len zeroed
// ---------------------------------------------------------------------------
__global__ void gather_embed_kernel(const int* __restrict__ tokens, int len,
                                    const float* __restrict__ emb,
                                    unsigned short* __restrict__ X) {
    int row = blockIdx.x;
    unsigned short* xr = X + (size_t)row * E_PAD;
    if (row < len) {
        int tok = tokens[row];
        const float* e = emb + (size_t)tok * E_DIM;
        for (int k = threadIdx.x; k < E_PAD; k += 128)
            xr[k] = (k < E_DIM) ? f2bf(e[k]) : (unsigned short)0;
    } else {
        for (int k = threadIdx.x; k < E_PAD; k += 128) xr[k] = 0;
    }
}

// ---------------------------------------------------------------------------
// Row-wise f32 -> bf16 with K padding (weights)
// ---------------------------------------------------------------------------
__global__ void convert_rows_kernel(const float* __restrict__ src,
                                    unsigned short* __restrict__ dst,
                                    int Ksrc, int Kdst) {
    int row = blockIdx.x;
    const float* s = src + (size_t)row * Ksrc;
    unsigned short* d = dst + (size_t)row * Kdst;
    for (int k = threadIdx.x; k < Kdst; k += 128)
        d[k] = (k < Ksrc) ? f2bf(s[k]) : (unsigned short)0;
}

// ---------------------------------------------------------------------------
// WMMA bf16 GEMM: C[M,N] = A[M,K] * B[N,K]^T (+bias[N]) (+relu)
// Block: 256 threads = 8 waves in 2(M) x 4(N) grid, each wave 32x32 tile.
// Grid: (N/128, M/64, batch). ldc = gridDim.x*128. K multiple of 32.
// ---------------------------------------------------------------------------
__device__ __forceinline__ v16bf load_frag(const unsigned short* p) {
    union { v16bf v; v8bf h[2]; } f;
    f.h[0] = *(const v8bf*)(p);        // K offsets sel*8 + 0..7
    f.h[1] = *(const v8bf*)(p + 16);   // K offsets 16 + sel*8 + 0..7
    return f.v;
}

__global__ __launch_bounds__(256)
void wmma_gemm_bf16_kernel(const unsigned short* __restrict__ A,
                           const unsigned short* __restrict__ B,
                           const float* __restrict__ bias,
                           float* __restrict__ C, int K,
                           long long sA, long long sB, long long sC,
                           long long sBias, int relu) {
    int z = blockIdx.z;
    const unsigned short* Ab = A + (size_t)z * sA;
    const unsigned short* Bb = B + (size_t)z * sB;
    float* Cb = C + (size_t)z * sC;
    const float* bp = bias ? (bias + (size_t)z * sBias) : nullptr;

    int ldc  = gridDim.x * 128;
    int lane = threadIdx.x & 31;
    int wid  = threadIdx.x >> 5;
    int m0   = blockIdx.y * 64  + (wid >> 2) * 32;
    int n0   = blockIdx.x * 128 + (wid & 3)  * 32;
    int r    = lane & 15;
    int sel  = lane >> 4;

    const unsigned short* pa0 = Ab + (size_t)(m0 + r)      * K + sel * 8;
    const unsigned short* pa1 = Ab + (size_t)(m0 + 16 + r) * K + sel * 8;
    const unsigned short* pb0 = Bb + (size_t)(n0 + r)      * K + sel * 8;
    const unsigned short* pb1 = Bb + (size_t)(n0 + 16 + r) * K + sel * 8;

    v8f acc00 = {}, acc01 = {}, acc10 = {}, acc11 = {};

    for (int k = 0; k < K; k += 32) {
        __builtin_prefetch(pa0 + k + 32, 0, 3);
        __builtin_prefetch(pb0 + k + 32, 0, 3);
        v16bf a0 = load_frag(pa0 + k);
        v16bf a1 = load_frag(pa1 + k);
        v16bf b0 = load_frag(pb0 + k);
        v16bf b1 = load_frag(pb1 + k);
        acc00 = __builtin_amdgcn_wmma_f32_16x16x32_bf16(false, a0, false, b0, (short)0, acc00, false, false);
        acc01 = __builtin_amdgcn_wmma_f32_16x16x32_bf16(false, a0, false, b1, (short)0, acc01, false, false);
        acc10 = __builtin_amdgcn_wmma_f32_16x16x32_bf16(false, a1, false, b0, (short)0, acc10, false, false);
        acc11 = __builtin_amdgcn_wmma_f32_16x16x32_bf16(false, a1, false, b1, (short)0, acc11, false, false);
    }

    // C/D layout: lanes 0-15 -> N=lane, M=vgpr; lanes 16-31 -> N=lane-16, M=vgpr+8
    v8f accs[2][2] = {{acc00, acc01}, {acc10, acc11}};
    #pragma unroll
    for (int mi = 0; mi < 2; ++mi) {
        #pragma unroll
        for (int nj = 0; nj < 2; ++nj) {
            int col  = n0 + nj * 16 + (lane & 15);
            int rowb = m0 + mi * 16 + sel * 8;
            float bv = bp ? bp[col] : 0.f;
            #pragma unroll
            for (int rr = 0; rr < 8; ++rr) {
                float v = accs[mi][nj][rr] + bv;
                if (relu) v = fmaxf(v, 0.f);
                Cb[(size_t)(rowb + rr) * ldc + col] = v;
            }
        }
    }
}

// ---------------------------------------------------------------------------
// Multi-WGP text LSTM scan. Grid = 2*NWG blocks, 256 threads.
// Workgroup w of direction d owns hidden units [32w, 32w+32): its 128 Whh
// rows (4 gate classes x 32 units, 128x512 f32 = 256KB) are preloaded into
// LDS once via global_load_async_to_lds_b128 and stay resident (CDNA5 320KB
// LDS). Per step: dot products from LDS, elementwise update for 32 units,
// h slice published to a double-buffered global h, grid-wide arrival-counter
// sync, reload h. Eliminates the per-step 4MB Whh L2 re-read entirely.
// ---------------------------------------------------------------------------
__global__ __launch_bounds__(256)
void lstm_scan_multi_kernel(const float* __restrict__ xp,      // [2, T, 4H]
                            const float* __restrict__ whh,     // [2, 4H, H]
                            const float* __restrict__ h0,
                            const float* __restrict__ c0, int T,
                            float* __restrict__ out_f32,       // [T, 2H]
                            unsigned short* __restrict__ out_bf16,
                            float* __restrict__ hT,            // [2H]
                            float* __restrict__ hbuf,          // [2 dir][2 slot][H]
                            int* __restrict__ sync_cnt) {      // [2]
    extern __shared__ float smem[];
    float* Wl  = smem;                  // 128 * 512 floats (256KB)
    float* h_s = Wl + 128 * HID;        // 512
    float* g_l = h_s + HID;             // 128
    float* c_l = g_l + 128;             // 32

    int dir = blockIdx.x / NWG;
    int w   = blockIdx.x % NWG;
    int tid = threadIdx.x;
    const float* W   = whh + (size_t)dir * G4 * HID;
    const float* xpd = xp + (size_t)dir * T * G4;

    // Async preload of this WG's Whh slice into LDS.
    // Local row r = gc*32+u  <->  global row gc*512 + w*32 + u.
    for (int idx = tid; idx < 128 * HID / 4; idx += 256) {
        int r  = idx >> 7;          // / (512/4)
        int k4 = idx & 127;
        int gr = (r >> 5) * HID + w * 32 + (r & 31);
        const float* g = W + (size_t)gr * HID + k4 * 4;
        unsigned lds = (unsigned)(size_t)(const void*)(Wl + (size_t)r * HID + k4 * 4);
        async_load_b128(lds, g);
    }
    for (int k = tid; k < HID; k += 256) h_s[k] = h0[dir * HID + k];
    if (tid < 32) c_l[tid] = c0[dir * HID + w * 32 + tid];
    wait_async0();
    __syncthreads();

    int rloc = tid >> 1;            // local gate row 0..127
    int half = tid & 1;             // which 256-wide K half
    const float* wr = Wl + (size_t)rloc * HID + half * 256;
    float hlast = h0[dir * HID + w * 32 + (tid & 31)];
    long spin_budget = 10000000;    // total bound: cannot hang graph replay

    for (int s = 0; s < T; ++s) {
        int t = dir ? (T - 1 - s) : s;
        const float* hh = h_s + half * 256;
        float d = 0.f;
        #pragma unroll 8
        for (int k = 0; k < 256; ++k) d += wr[k] * hh[k];
        d += __shfl_xor(d, 1, 32);                  // combine K halves
        if (half == 0) g_l[rloc] = d;
        __syncthreads();

        int slot = (s + 1) & 1;
        if (tid < 32) {
            int j = w * 32 + tid;
            const float* xpt = xpd + (size_t)t * G4;
            float ig = sigf(xpt[j] + g_l[tid]);
            float fg = sigf(xpt[HID + j] + g_l[32 + tid]);
            float gg = tanhf(xpt[2 * HID + j] + g_l[64 + tid]);
            float og = sigf(xpt[3 * HID + j] + g_l[96 + tid]);
            float c  = fg * c_l[tid] + ig * gg;
            float h  = og * tanhf(c);
            c_l[tid] = c;
            hlast = h;
            hbuf[(dir * 2 + slot) * HID + j] = h;
            if (out_f32)  out_f32[(size_t)t * H2 + dir * HID + j] = h;
            if (out_bf16) out_bf16[(size_t)t * H2 + dir * HID + j] = f2bf(h);
        }
        __threadfence();
        __syncthreads();
        if (tid == 0) {
            atomicAdd(&sync_cnt[dir], 1);
            int target = NWG * (s + 1);
            volatile int* vc = (volatile int*)(sync_cnt + dir);
            while (*vc < target && spin_budget > 0) --spin_budget;
        }
        __syncthreads();
        const volatile float* hb = (const volatile float*)(hbuf + (dir * 2 + slot) * HID);
        for (int k = tid; k < HID; k += 256) h_s[k] = hb[k];
        __syncthreads();
    }
    if (tid < 32) hT[dir * HID + w * 32 + tid] = hlast;
}

__global__ void init_sync_kernel(int* __restrict__ sync_cnt,
                                 float* __restrict__ hbuf) {
    int tid = threadIdx.x;
    if (tid < 2) sync_cnt[tid] = 0;
    for (int k = tid; k < 2 * 2 * HID; k += 256) hbuf[k] = 0.f;
}

// ---------------------------------------------------------------------------
// Single-WG bidirectional LSTM scan for short sequences (q/a).
// Grid = 2 blocks (dir), 1024 threads each.
// ---------------------------------------------------------------------------
__global__ __launch_bounds__(1024)
void lstm_scan_kernel(const float* __restrict__ xp, int xpRows,
                      const float* __restrict__ whh,
                      const float* __restrict__ h0, const float* __restrict__ c0,
                      int T, float* __restrict__ hT) {
    __shared__ __align__(16) float h_s[HID];
    __shared__ __align__(16) float g_s[G4];
    __shared__ __align__(16) float c_s[HID];

    int dir = blockIdx.x;
    int tid = threadIdx.x;
    const float* xpd = xp + (size_t)dir * xpRows * G4;
    const float* W   = whh + (size_t)dir * G4 * HID;

    if (tid < HID) { h_s[tid] = h0[dir * HID + tid]; c_s[tid] = c0[dir * HID + tid]; }
    __syncthreads();

    int ga = tid, gb = tid + 1024;
    const float4* Wa = (const float4*)(W + (size_t)ga * HID);
    const float4* Wb = (const float4*)(W + (size_t)gb * HID);

    for (int s = 0; s < T; ++s) {
        int t = dir ? (T - 1 - s) : s;
        const float* xpt = xpd + (size_t)t * G4;
        const float4* hv = (const float4*)h_s;
        float da = 0.f, db = 0.f;
        #pragma unroll 8
        for (int k = 0; k < HID / 4; ++k) {
            float4 h4 = hv[k];
            float4 wa = Wa[k];
            float4 wb = Wb[k];
            da += wa.x * h4.x + wa.y * h4.y + wa.z * h4.z + wa.w * h4.w;
            db += wb.x * h4.x + wb.y * h4.y + wb.z * h4.z + wb.w * h4.w;
        }
        g_s[ga] = xpt[ga] + da;
        g_s[gb] = xpt[gb] + db;
        __syncthreads();
        if (tid < HID) {
            float ig = sigf(g_s[tid]);
            float fg = sigf(g_s[HID + tid]);
            float gg = tanhf(g_s[2 * HID + tid]);
            float og = sigf(g_s[3 * HID + tid]);
            float c  = fg * c_s[tid] + ig * gg;
            float h  = og * tanhf(c);
            c_s[tid] = c;
            h_s[tid] = h;
        }
        __syncthreads();
    }
    if (tid < HID) hT[dir * HID + tid] = h_s[tid];
}

// ---------------------------------------------------------------------------
// qa = relu(att_w @ h_a + att_b)   (512 outputs, K=1024). 1 block, 512 thr.
// ---------------------------------------------------------------------------
__global__ __launch_bounds__(512)
void qa_vec_kernel(const float* __restrict__ att_w, const float* __restrict__ att_b,
                   const float* __restrict__ h_a, float* __restrict__ qa) {
    __shared__ float ha_s[H2];
    int tid = threadIdx.x;
    ha_s[tid]       = h_a[tid];
    ha_s[tid + 512] = h_a[tid + 512];
    __syncthreads();
    const float* w = att_w + (size_t)tid * H2;
    float s = att_b[tid];
    #pragma unroll 4
    for (int k = 0; k < H2; ++k) s += w[k] * ha_s[k];
    qa[tid] = fmaxf(s, 0.f);
}

// ---------------------------------------------------------------------------
// scores = keys @ qa; alphas = softmax(scores); att = alphas @ out_t
// ---------------------------------------------------------------------------
__global__ __launch_bounds__(1024)
void attend_kernel(const float* __restrict__ keys, const float* __restrict__ qa,
                   const float* __restrict__ out_t, float* __restrict__ att) {
    __shared__ __align__(16) float qa_s[HID];
    __shared__ float sc[T_LEN];
    __shared__ float red[1024];
    int tid = threadIdx.x;
    if (tid < HID) qa_s[tid] = qa[tid];
    __syncthreads();

    for (int t = tid; t < T_LEN; t += 1024) {
        const float* kr = keys + (size_t)t * HID;
        float s = 0.f;
        #pragma unroll 4
        for (int k = 0; k < HID; ++k) s += kr[k] * qa_s[k];
        sc[t] = s;
    }
    __syncthreads();

    red[tid] = fmaxf(sc[tid], sc[tid + 1024]);
    __syncthreads();
    for (int off = 512; off > 0; off >>= 1) {
        if (tid < off) red[tid] = fmaxf(red[tid], red[tid + off]);
        __syncthreads();
    }
    float mx = red[0];
    __syncthreads();

    float e0 = __expf(sc[tid] - mx);
    float e1 = __expf(sc[tid + 1024] - mx);
    sc[tid] = e0;
    sc[tid + 1024] = e1;
    red[tid] = e0 + e1;
    __syncthreads();
    for (int off = 512; off > 0; off >>= 1) {
        if (tid < off) red[tid] += red[tid + off];
        __syncthreads();
    }
    float inv = 1.f / red[0];

    float a = 0.f;
    for (int t = 0; t < T_LEN; ++t) a += sc[t] * out_t[(size_t)t * H2 + tid];
    att[tid] = a * inv;
}

// ---------------------------------------------------------------------------
// s1 = att . (W1 @ h_a); s2 = h_q . (W2 @ h_a); out = sigmoid(s1+s2+b1+b2)
// ---------------------------------------------------------------------------
__global__ __launch_bounds__(1024)
void bilinear_kernel(const float* __restrict__ att, const float* __restrict__ h_q,
                     const float* __restrict__ h_a,
                     const float* __restrict__ w1, const float* __restrict__ b1,
                     const float* __restrict__ w2, const float* __restrict__ b2,
                     float* __restrict__ out) {
    __shared__ float ha_s[H2];
    __shared__ float red[1024];
    int tid = threadIdx.x;
    ha_s[tid] = h_a[tid];
    __syncthreads();
    const float* r1 = w1 + (size_t)tid * H2;
    const float* r2 = w2 + (size_t)tid * H2;
    float v1 = 0.f, v2 = 0.f;
    #pragma unroll 4
    for (int k = 0; k < H2; ++k) { v1 += r1[k] * ha_s[k]; v2 += r2[k] * ha_s[k]; }
    red[tid] = att[tid] * v1 + h_q[tid] * v2;
    __syncthreads();
    for (int off = 512; off > 0; off >>= 1) {
        if (tid < off) red[tid] += red[tid + off];
        __syncthreads();
    }
    if (tid == 0) {
        float s = red[0] + b1[0] + b2[0];
        out[0] = 1.f / (1.f + __expf(-s));
    }
}

// ---------------------------------------------------------------------------
extern "C" void kernel_launch(void* const* d_in, const int* in_sizes, int n_in,
                              void* d_out, int out_size, void* d_ws, size_t ws_size,
                              hipStream_t stream) {
    (void)in_sizes; (void)n_in; (void)out_size; (void)ws_size;

    const int*   text     = (const int*)d_in[0];
    const int*   question = (const int*)d_in[1];
    const int*   answer   = (const int*)d_in[2];
    const float* emb      = (const float*)d_in[3];
    const float* t_wih = (const float*)d_in[4];
    const float* t_whh = (const float*)d_in[5];
    const float* t_b   = (const float*)d_in[6];
    const float* h0_t  = (const float*)d_in[7];
    const float* c0_t  = (const float*)d_in[8];
    const float* q_wih = (const float*)d_in[9];
    const float* q_whh = (const float*)d_in[10];
    const float* q_b   = (const float*)d_in[11];
    const float* h0_q  = (const float*)d_in[12];
    const float* c0_q  = (const float*)d_in[13];
    const float* a_wih = (const float*)d_in[14];
    const float* a_whh = (const float*)d_in[15];
    const float* a_b   = (const float*)d_in[16];
    const float* h0_a  = (const float*)d_in[17];
    const float* c0_a  = (const float*)d_in[18];
    const float* att_w  = (const float*)d_in[19];
    const float* att_b  = (const float*)d_in[20];
    const float* bil1_w = (const float*)d_in[21];
    const float* bil1_b = (const float*)d_in[22];
    const float* bil2_w = (const float*)d_in[23];
    const float* bil2_b = (const float*)d_in[24];

    char* ws = (char*)d_ws;
    size_t off = 0;
    auto alloc = [&](size_t bytes) -> char* {
        char* p = ws + off;
        off = (off + bytes + 255) & ~(size_t)255;
        return p;
    };

    unsigned short* Xt    = (unsigned short*)alloc((size_t)T_LEN * E_PAD * 2);
    unsigned short* Xq    = (unsigned short*)alloc((size_t)Q_LEN * E_PAD * 2);
    unsigned short* Xa    = (unsigned short*)alloc((size_t)64    * E_PAD * 2);
    unsigned short* Wt    = (unsigned short*)alloc((size_t)2 * G4 * E_PAD * 2);
    unsigned short* Wq    = (unsigned short*)alloc((size_t)2 * G4 * E_PAD * 2);
    unsigned short* Wa    = (unsigned short*)alloc((size_t)2 * G4 * E_PAD * 2);
    unsigned short* AttWb = (unsigned short*)alloc((size_t)HID * H2 * 2);
    float* xp_t   = (float*)alloc((size_t)2 * T_LEN * G4 * 4);
    float* xp_q   = (float*)alloc((size_t)2 * 64 * G4 * 4);
    float* xp_a   = (float*)alloc((size_t)2 * 64 * G4 * 4);
    float* outT   = (float*)alloc((size_t)T_LEN * H2 * 4);
    unsigned short* outTb = (unsigned short*)alloc((size_t)T_LEN * H2 * 2);
    float* hT_t   = (float*)alloc((size_t)H2 * 4);
    float* hT_q   = (float*)alloc((size_t)H2 * 4);
    float* hT_a   = (float*)alloc((size_t)H2 * 4);
    float* keys   = (float*)alloc((size_t)T_LEN * HID * 4);
    float* qa     = (float*)alloc((size_t)HID * 4);
    float* attv   = (float*)alloc((size_t)H2 * 4);
    float* hbuf   = (float*)alloc((size_t)2 * 2 * HID * 4);
    int*   syncp  = (int*)alloc(256);

    // 1) Gather embeddings -> bf16 (K padded to 320); convert weights -> bf16
    gather_embed_kernel<<<T_LEN, 128, 0, stream>>>(text, T_LEN, emb, Xt);
    gather_embed_kernel<<<Q_LEN, 128, 0, stream>>>(question, Q_LEN, emb, Xq);
    gather_embed_kernel<<<64,    128, 0, stream>>>(answer, A_LEN, emb, Xa);
    convert_rows_kernel<<<2 * G4, 128, 0, stream>>>(t_wih, Wt, E_DIM, E_PAD);
    convert_rows_kernel<<<2 * G4, 128, 0, stream>>>(q_wih, Wq, E_DIM, E_PAD);
    convert_rows_kernel<<<2 * G4, 128, 0, stream>>>(a_wih, Wa, E_DIM, E_PAD);
    convert_rows_kernel<<<HID, 128, 0, stream>>>(att_w, AttWb, H2, H2);
    init_sync_kernel<<<1, 256, 0, stream>>>(syncp, hbuf);

    // 2) Input projections xp = X @ Wih^T + b  (WMMA, batched over direction)
    dim3 blk(256);
    wmma_gemm_bf16_kernel<<<dim3(G4 / 128, T_LEN / 64, 2), blk, 0, stream>>>(
        Xt, Wt, t_b, xp_t, E_PAD, 0LL, (long long)G4 * E_PAD,
        (long long)T_LEN * G4, (long long)G4, 0);
    wmma_gemm_bf16_kernel<<<dim3(G4 / 128, 1, 2), blk, 0, stream>>>(
        Xq, Wq, q_b, xp_q, E_PAD, 0LL, (long long)G4 * E_PAD,
        (long long)64 * G4, (long long)G4, 0);
    wmma_gemm_bf16_kernel<<<dim3(G4 / 128, 1, 2), blk, 0, stream>>>(
        Xa, Wa, a_b, xp_a, E_PAD, 0LL, (long long)G4 * E_PAD,
        (long long)64 * G4, (long long)G4, 0);

    // 3) Recurrent scans. Text: 32 WGPs, Whh resident in LDS (async preload).
    size_t scan_lds = (128 * HID + HID + 128 + 32) * sizeof(float);  // ~259KB
    lstm_scan_multi_kernel<<<2 * NWG, 256, scan_lds, stream>>>(
        xp_t, t_whh, h0_t, c0_t, T_LEN, outT, outTb, hT_t, hbuf, syncp);
    lstm_scan_kernel<<<2, 1024, 0, stream>>>(xp_q, 64, q_whh, h0_q, c0_q,
                                             Q_LEN, hT_q);
    lstm_scan_kernel<<<2, 1024, 0, stream>>>(xp_a, 64, a_whh, h0_a, c0_a,
                                             A_LEN, hT_a);

    // 4) Attention: qa vector, keys GEMM (WMMA, fused relu), softmax + weighted sum
    qa_vec_kernel<<<1, 512, 0, stream>>>(att_w, att_b, hT_a, qa);
    wmma_gemm_bf16_kernel<<<dim3(HID / 128, T_LEN / 64, 1), blk, 0, stream>>>(
        outTb, AttWb, att_b, keys, H2, 0LL, 0LL, 0LL, 0LL, 1);
    attend_kernel<<<1, 1024, 0, stream>>>(keys, qa, outT, attv);

    // 5) Bilinear scores + sigmoid
    bilinear_kernel<<<1, 1024, 0, stream>>>(attv, hT_q, hT_a, bil1_w, bil1_b,
                                            bil2_w, bil2_b, (float*)d_out);
}
